// ResConvModule_5317169512856
// MI455X (gfx1250) — matrix operand order repaired and checked by reading, not compiled, not run on hardware
//
#include <hip/hip_runtime.h>
#include <hip/hip_bf16.h>
#include <stdint.h>

// ---------------------------------------------------------------------------
// MI455X (gfx1250) implementation.
//  * wave32; WMMA f32_16x16x32_f16 for all ResNet convs (fp32 accum)
//  * im2col K ordered tap-major (K = dt*CIN + c): per-32-chunk dt is uniform,
//    so zero-padding is a uniform chunk skip and A fragments are two
//    contiguous ds_load_b128 per lane (activations stored time-major in LDS)
//  * weights pre-packed into WMMA B-fragment layout (fp16) in workspace (L2)
//  * Tensor Data Mover (tensor_load_to_lds + s_wait_tensorcnt) stages the
//    per-block activation tile global->LDS (6-arg builtin on this toolchain)
// ---------------------------------------------------------------------------

typedef __attribute__((ext_vector_type(16))) _Float16 v16h;
typedef __attribute__((ext_vector_type(8)))  _Float16 v8h;
typedef __attribute__((ext_vector_type(8)))  float    v8f;

// ---------------- model constants -------------------------------------------
#define BATCH   16384
#define LPEP    21
#define MMHC    34
#define EDIM    16
#define T0      7
#define C0      96
#define XFE_ELT 672          // 96 * 7 halves per sample (time-major: [t][c])

// ResNet conv table (14 convs, packing order)
constexpr int kNConv = 14;
constexpr int kConvCin[kNConv]  = {96,96,96,96, 96,128,96, 128,128, 128,256,128, 256,256};
constexpr int kConvKW [kNConv]  = { 3, 3, 3, 3,  3,  3, 1,   3,  3,   3,  3,  1,   3,  3};
// frags per conv = (Cout/16) * (Cin*kW/32); prefix sums:
constexpr int kConvOff[kNConv+1] = {0,54,108,162,216,288,384,408,504,600,792,1176,1240,1624,2008};
constexpr int kTotalFrags = 2008;                    // * 512 halves each

// ---------------- kernel arg structs ----------------------------------------
struct PackArgs  { const float* w[kNConv]; _Float16* wpack; };

struct FrontArgs {
  const int*   pep_x;  const int*   mhc_x;
  const float* pep_emb; const float* mhc_emb;
  const float* attn_in_w; const float* attn_in_b;
  const float* attn_out_w; const float* attn_out_b;
  const float* cconv_w; const float* cconv_b;
  const float* cln_w;   const float* cln_b;
  const float* iconv_w[4]; const float* iconv_b[4];
  const float* iln_w[4];   const float* iln_b[4];
  _Float16* xfe;
};

struct ResArgs {
  const _Float16* xfe;
  const _Float16* wpack;
  const float* ln_g[kNConv];
  const float* ln_b[kNConv];
  float* out;
};

// ===========================================================================
// 1) weight packing: fp32 (Cout, Cin, kW) -> fp16 B-fragments with
//    K = dt*CIN + c ordering.
//    B (32x16 f16): lanes 0-15 N=lane, K=h<8?h:h+8; lanes 16-31 N=lane-16, K=8+(h<8?h:h+8)
// ===========================================================================
__global__ __launch_bounds__(32) void pack_weights_kernel(PackArgs pa) {
  int f = blockIdx.x;
  int ci = 0;
  while (ci < kNConv - 1 && f >= kConvOff[ci + 1]) ci++;
  const int fl   = f - kConvOff[ci];
  const int Cin  = kConvCin[ci];
  const int kW   = kConvKW[ci];
  const int kc   = (Cin * kW) >> 5;            // Cin*kW is a multiple of 32
  const int ccI  = fl / kc;
  const int kch  = fl - ccI * kc;
  const int lane = threadIdx.x;
  const int cout = ccI * 16 + (lane & 15);
  const float* w = pa.w[ci];
  _Float16* dst  = pa.wpack + (size_t)f * 512 + lane * 16;
  const int kb   = kch * 32 + ((lane >> 4) << 3);
#pragma unroll
  for (int h = 0; h < 16; ++h) {
    int kk  = kb + h + (h >= 8 ? 8 : 0); // K = dt*Cin + c
    int dt  = kk / Cin;
    int c   = kk - dt * Cin;
    dst[h]  = (_Float16)w[((size_t)cout * Cin + c) * kW + dt];
  }
}

// ===========================================================================
// 2) frontend: one wave per sample -> fp16 (7,96) time-major concat tensor
// ===========================================================================
__global__ __launch_bounds__(32) void frontend_kernel(FrontArgs fa) {
  __shared__ float pepS[LPEP][EDIM];
  __shared__ float mhcS[MMHC][EDIM];
  __shared__ float qS[15][EDIM];
  __shared__ float kS[MMHC][EDIM];
  __shared__ float vS[MMHC][EDIM];
  __shared__ float oS[15][EDIM];
  __shared__ float scS[4][15][MMHC];
  __shared__ float kernS[16 * 16 * 15];
  __shared__ float xc[C0][T0];

  const int lane = threadIdx.x;
  const int s    = blockIdx.x;

  // embedding gathers
  for (int i = lane; i < LPEP * EDIM; i += 32) {
    int t = i >> 4, e = i & 15;
    pepS[t][e] = fa.pep_emb[fa.pep_x[s * LPEP + t] * EDIM + e];
  }
  for (int i = lane; i < MMHC * EDIM; i += 32) {
    int t = i >> 4, e = i & 15;
    mhcS[t][e] = fa.mhc_emb[fa.mhc_x[s * MMHC + t] * EDIM + e];
  }
  __syncthreads();

  // q / k / v projections
  for (int i = lane; i < 15 * EDIM; i += 32) {
    int t = i >> 4, e = i & 15;
    float a = fa.attn_in_b[e];
    for (int c = 0; c < EDIM; ++c) a += pepS[t + 3][c] * fa.attn_in_w[e * EDIM + c];
    qS[t][e] = a;
  }
  for (int i = lane; i < MMHC * EDIM; i += 32) {
    int t = i >> 4, e = i & 15;
    float ak = fa.attn_in_b[16 + e], av = fa.attn_in_b[32 + e];
    for (int c = 0; c < EDIM; ++c) {
      ak += mhcS[t][c] * fa.attn_in_w[(16 + e) * EDIM + c];
      av += mhcS[t][c] * fa.attn_in_w[(32 + e) * EDIM + c];
    }
    kS[t][e] = ak;  vS[t][e] = av;
  }
  __syncthreads();

  // scores (H=4, D=4, scale 1/sqrt(4)=0.5)
  for (int i = lane; i < 4 * 15 * MMHC; i += 32) {
    int h2 = i / (15 * MMHC);
    int r  = i - h2 * 15 * MMHC;
    int qt = r / MMHC, kt = r - qt * MMHC;
    float a = 0.f;
    for (int d = 0; d < 4; ++d) a += qS[qt][h2 * 4 + d] * kS[kt][h2 * 4 + d];
    scS[h2][qt][kt] = a * 0.5f;
  }
  __syncthreads();

  // softmax over k (34)
  for (int i = lane; i < 4 * 15; i += 32) {
    int h2 = i / 15, qt = i - h2 * 15;
    float mx = -1e30f;
    for (int kt = 0; kt < MMHC; ++kt) mx = fmaxf(mx, scS[h2][qt][kt]);
    float sum = 0.f;
    for (int kt = 0; kt < MMHC; ++kt) { float e = __expf(scS[h2][qt][kt] - mx); scS[h2][qt][kt] = e; sum += e; }
    float inv = 1.f / sum;
    for (int kt = 0; kt < MMHC; ++kt) scS[h2][qt][kt] *= inv;
  }
  __syncthreads();

  // attn output (into oS) then output projection (into qS)
  for (int i = lane; i < 15 * EDIM; i += 32) {
    int t = i >> 4, e = i & 15;
    int h2 = e >> 2;
    float a = 0.f;
    for (int kt = 0; kt < MMHC; ++kt) a += scS[h2][t][kt] * vS[kt][e];
    oS[t][e] = a;
  }
  __syncthreads();
  for (int i = lane; i < 15 * EDIM; i += 32) {
    int t = i >> 4, e = i & 15;
    float a = fa.attn_out_b[e];
    for (int c = 0; c < EDIM; ++c) a += oS[t][c] * fa.attn_out_w[e * EDIM + c];
    qS[t][e] = a;
  }
  __syncthreads();

  // cconv (32ch, k=9, valid) + relu -> channels 64..95 of xc
  for (int i = lane; i < 32 * T0; i += 32) {
    int oc = i / T0, t = i - oc * T0;
    float a = fa.cconv_b[oc];
    for (int ic = 0; ic < EDIM; ++ic)
      for (int j = 0; j < 9; ++j)
        a += qS[t + j][ic] * fa.cconv_w[(oc * EDIM + ic) * 9 + j];
    xc[64 + oc][t] = fmaxf(a, 0.f);
  }
  __syncthreads();
  for (int t = lane; t < T0; t += 32) {
    float sum = 0.f, sq = 0.f;
    for (int oc = 0; oc < 32; ++oc) { float x = xc[64 + oc][t]; sum += x; sq += x * x; }
    float m = sum / 32.f, var = sq / 32.f - m * m, inv = rsqrtf(var + 1e-5f);
    for (int oc = 0; oc < 32; ++oc) {
      float x = xc[64 + oc][t];
      xc[64 + oc][t] = (x - m) * inv * fa.cln_w[oc] + fa.cln_b[oc];
    }
  }
  __syncthreads();

  // interaction branches
  const int csA[4]  = {9, 11, 13, 15};
  const int offA[4] = {3, 2, 1, 0};
  for (int br = 0; br < 4; ++br) {
    const int cs = csA[br], off = offA[br];
    const float* w = fa.iconv_w[br];
    // data-dependent kernel: kern[o][d][k] = relu(sum_l mhc[l][d]*w[o][k][l])
    for (int idx = lane; idx < 16 * 16 * cs; idx += 32) {
      int o = idx / (16 * cs);
      int r = idx - o * 16 * cs;
      int d = r / cs, k = r - d * cs;
      float a = 0.f;
      for (int l = 0; l < MMHC; ++l) a += mhcS[l][d] * w[(o * cs + k) * MMHC + l];
      kernS[(o * 16 + d) * 15 + k] = fmaxf(a, 0.f);
    }
    __syncthreads();
    // y[o][t] = sum_{k,d} pep[off+t+k][d] * kern[o][d][k] + b
    for (int idx = lane; idx < 16 * T0; idx += 32) {
      int o = idx / T0, t = idx - o * T0;
      float a = fa.iconv_b[br][o];
      for (int k = 0; k < cs; ++k)
        for (int d = 0; d < EDIM; ++d)
          a += pepS[off + t + k][d] * kernS[(o * 16 + d) * 15 + k];
      xc[br * 16 + o][t] = fmaxf(a, 0.f);
    }
    __syncthreads();
    for (int t = lane; t < T0; t += 32) {
      float sum = 0.f, sq = 0.f;
      for (int o = 0; o < 16; ++o) { float x = xc[br * 16 + o][t]; sum += x; sq += x * x; }
      float m = sum / 16.f, var = sq / 16.f - m * m, inv = rsqrtf(var + 1e-5f);
      for (int o = 0; o < 16; ++o) {
        float x = xc[br * 16 + o][t];
        xc[br * 16 + o][t] = (x - m) * inv * fa.iln_w[br][o] + fa.iln_b[br][o];
      }
    }
    __syncthreads();
  }

  // emit fp16 per sample, time-major: xfe[s][t*96 + c]
  _Float16* dst = fa.xfe + (size_t)s * XFE_ELT;
  for (int i = lane; i < XFE_ELT; i += 32) {
    int t = i / C0, c = i - t * C0;
    dst[i] = (_Float16)xc[c][t];
  }
}

// ===========================================================================
// 3) ResNet: 16-sample tiles, WMMA GEMM convs, block-cooperative LN
//    activation layout (time-major): buf[s][t*C + c]
// ===========================================================================
#define RS_BT 128   // 4 wave32s

template <int CIN, int COUT, int KW, int STRIDE, int TIN, int TOUT>
__device__ __forceinline__ void conv_wmma(const _Float16* __restrict__ wp,
                                          const _Float16 (*in)[672],
                                          _Float16 (*out)[672]) {
  constexpr int KC = (CIN * KW) / 32;   // CIN is a multiple of 32
  constexpr int CC = COUT / 16;
  const int lane = threadIdx.x & 31;
  const int wv   = threadIdx.x >> 5;
  const int mrow = lane & 15;          // A row = sample
  const int hi   = lane >> 4;
  const v8f zf = {};

  for (int t = wv; t < TOUT; t += 4) {   // whole-wave t loop: EXEC stays full
    v8f acc[CC];
#pragma unroll
    for (int i = 0; i < CC; ++i) acc[i] = zf;

    for (int kch = 0; kch < KC; ++kch) {
      // K = dt*CIN + c  ->  dt (and tt) are uniform across the chunk
      const int dt = (KW == 1) ? 0 : ((kch * 32) / CIN);
      const int cb = kch * 32 - dt * CIN;
      const int tt = t * STRIDE + dt - (KW == 3 ? 1 : 0);
      if (KW == 3 && (tt < 0 || tt >= TIN)) continue;   // uniform zero-pad skip

      // A (16x32 f16): lane row = sample; halves 0..7 = K cb+hi*8..+7,
      // halves 8..15 = K cb+16+hi*8..+7  -> two contiguous b128 LDS loads
      const _Float16* base = &in[mrow][tt * CIN + cb + (hi << 3)];
      v8h alo = *(const v8h*)(base);
      v8h ahi = *(const v8h*)(base + 16);
      v16h a;
#pragma unroll
      for (int h = 0; h < 8; ++h) { a[h] = alo[h]; a[h + 8] = ahi[h]; }

#pragma unroll
      for (int ci = 0; ci < CC; ++ci) {
        v16h b = *(const v16h*)(wp + ((size_t)(ci * KC + kch) * 512) + lane * 16);
        acc[ci] = __builtin_amdgcn_wmma_f32_16x16x32_f16(
            false, a, false, b, (short)0, acc[ci], false, false);
      }
    }
    // D layout: lane&15 = N(cout), VGPR j = M row j (+8 for lanes>=16)
#pragma unroll
    for (int ci = 0; ci < CC; ++ci) {
      int n = lane & 15;
#pragma unroll
      for (int j = 0; j < 8; ++j) {
        int s = j + (hi << 3);
        out[s][t * COUT + ci * 16 + n] = (_Float16)acc[ci][j];
      }
    }
  }
}

__device__ __forceinline__ void ln_ip(_Float16 (*buf)[672], int C, int T,
                                      const float* g, const float* bb, bool do_relu) {
  for (int p = threadIdx.x; p < 16 * T; p += RS_BT) {
    int s = p / T, t = p - (p / T) * T;
    float sum = 0.f, sq = 0.f;
    for (int c = 0; c < C; ++c) { float x = (float)buf[s][t * C + c]; sum += x; sq += x * x; }
    float m = sum / C, var = sq / C - m * m, inv = rsqrtf(var + 1e-5f);
    for (int c = 0; c < C; ++c) {
      float x = (float)buf[s][t * C + c];
      float y = (x - m) * inv * g[c] + bb[c];
      if (do_relu) y = fmaxf(y, 0.f);
      buf[s][t * C + c] = (_Float16)y;
    }
  }
}

__device__ __forceinline__ void combine_relu(_Float16 (*X)[672], const _Float16 (*O2)[672],
                                             const _Float16 (*R)[672], int C, int T) {
  const int n = 16 * C * T;
  for (int p = threadIdx.x; p < n; p += RS_BT) {
    int s = p / (C * T), e = p - s * (C * T);
    float y = (float)O2[s][e] + (float)R[s][e];
    X[s][e] = (_Float16)fmaxf(y, 0.f);
  }
}

__global__ __launch_bounds__(RS_BT) void resnet_kernel(ResArgs ra) {
  __shared__ __align__(16) _Float16 sb[3][16][672];   // 63 KB: X / U / W2
  _Float16 (*X)[672]  = sb[0];
  _Float16 (*U)[672]  = sb[1];
  _Float16 (*W2)[672] = sb[2];
  const int tid = threadIdx.x;

  __builtin_prefetch(ra.wpack, 0, 3);   // global_prefetch: warm packed weights

  // ---- stage the 16-sample tile (21504 B contiguous) into X ----
#if __has_builtin(__builtin_amdgcn_tensor_load_to_lds)
  {
    // Tensor Data Mover: 1-D tile, dword elements (ISA 08, D# groups 0/1).
    typedef __attribute__((ext_vector_type(4))) unsigned int u32x4;
    typedef __attribute__((ext_vector_type(8))) int          i32x8;
    typedef __attribute__((ext_vector_type(4))) int          i32x4;
    const unsigned long long ga =
        (unsigned long long)(uintptr_t)(ra.xfe + (size_t)blockIdx.x * 16 * XFE_ELT);
    // flat LDS address: low 32 bits are the LDS byte offset (aperture rule)
    const unsigned int ldsOff = (unsigned int)(uintptr_t)&sb[0][0][0];
    const unsigned int NDW = (16 * XFE_ELT) / 2;   // 5376 dwords

    u32x4 g0;
    g0[0] = 1u;                                        // count=1 (valid), user mode
    g0[1] = ldsOff;                                    // lds_addr (bytes)
    g0[2] = (unsigned int)(ga & 0xffffffffull);        // global_addr[31:0]
    g0[3] = (unsigned int)((ga >> 32) & 0x01ffffffull) // global_addr[56:32]
            | 0x80000000u;                             // type=2 ("image") @ [127:126]

    i32x8 g1;
    g1[0] = (int)(2u << 16);                 // data_size=2 -> 4-byte elements
    g1[1] = (int)((NDW & 0xffffu) << 16);    // tensor_dim0[15:0]
    g1[2] = (int)(((NDW >> 16) & 0xffffu)    // tensor_dim0[31:16]
            | (1u << 16));                   // tensor_dim1 = 1
    g1[3] = (int)((NDW & 0xffffu) << 16);    // tile_dim0 = 5376
    g1[4] = 0;                               // tile_dim1/2 unused
    g1[5] = (int)NDW;                        // tensor_dim0_stride[31:0]
    g1[6] = 0;                               // stride hi bits
    g1[7] = 0;
    i32x4 z4; z4[0] = 0; z4[1] = 0; z4[2] = 0; z4[3] = 0;
    i32x8 z8; z8[0] = 0; z8[1] = 0; z8[2] = 0; z8[3] = 0;
    z8[4] = 0; z8[5] = 0; z8[6] = 0; z8[7] = 0;

    // every wave issues the same DMA (TDM ignores EXEC; identical writes are
    // benign) and waits its own TENSORcnt before the workgroup barrier.
    __builtin_amdgcn_tensor_load_to_lds(g0, g1, z4, z4, z8, 0);
#if __has_builtin(__builtin_amdgcn_s_wait_tensorcnt)
    __builtin_amdgcn_s_wait_tensorcnt((short)0);
#endif
  }
#else
  {
    const uint32_t* src = (const uint32_t*)(ra.xfe + (size_t)blockIdx.x * 16 * XFE_ELT);
    uint32_t* dst = (uint32_t*)&sb[0][0][0];
    for (int i = tid; i < 16 * (XFE_ELT / 2); i += RS_BT) dst[i] = src[i];
  }
#endif
  __syncthreads();

#define WPTR(i) (ra.wpack + (size_t)kConvOff[i] * 512)
  // ---- l1 block 0 (96,7) ----
  conv_wmma<96,96,3,1,7,7>(WPTR(0), X, U);   __syncthreads();
  ln_ip(U, 96, 7, ra.ln_g[0], ra.ln_b[0], true);   __syncthreads();
  conv_wmma<96,96,3,1,7,7>(WPTR(1), U, W2);  __syncthreads();
  ln_ip(W2, 96, 7, ra.ln_g[1], ra.ln_b[1], false); __syncthreads();
  combine_relu(X, W2, X, 96, 7);             __syncthreads();
  // ---- l1 block 1 ----
  conv_wmma<96,96,3,1,7,7>(WPTR(2), X, U);   __syncthreads();
  ln_ip(U, 96, 7, ra.ln_g[2], ra.ln_b[2], true);   __syncthreads();
  conv_wmma<96,96,3,1,7,7>(WPTR(3), U, W2);  __syncthreads();
  ln_ip(W2, 96, 7, ra.ln_g[3], ra.ln_b[3], false); __syncthreads();
  combine_relu(X, W2, X, 96, 7);             __syncthreads();
  // ---- l2 block 0 (96->128, stride 2, T 7->4) ----
  conv_wmma<96,128,3,2,7,4>(WPTR(4), X, U);  __syncthreads();
  ln_ip(U, 128, 4, ra.ln_g[4], ra.ln_b[4], true);  __syncthreads();
  conv_wmma<128,128,3,1,4,4>(WPTR(5), U, W2);__syncthreads();
  ln_ip(W2, 128, 4, ra.ln_g[5], ra.ln_b[5], false);__syncthreads();
  conv_wmma<96,128,1,2,7,4>(WPTR(6), X, U);  __syncthreads();   // downsample
  ln_ip(U, 128, 4, ra.ln_g[6], ra.ln_b[6], false); __syncthreads();
  combine_relu(X, W2, U, 128, 4);            __syncthreads();
  // ---- l2 block 1 (128,4) ----
  conv_wmma<128,128,3,1,4,4>(WPTR(7), X, U); __syncthreads();
  ln_ip(U, 128, 4, ra.ln_g[7], ra.ln_b[7], true);  __syncthreads();
  conv_wmma<128,128,3,1,4,4>(WPTR(8), U, W2);__syncthreads();
  ln_ip(W2, 128, 4, ra.ln_g[8], ra.ln_b[8], false);__syncthreads();
  combine_relu(X, W2, X, 128, 4);            __syncthreads();
  // ---- l3 block 0 (128->256, stride 2, T 4->2) ----
  conv_wmma<128,256,3,2,4,2>(WPTR(9), X, U); __syncthreads();
  ln_ip(U, 256, 2, ra.ln_g[9], ra.ln_b[9], true);  __syncthreads();
  conv_wmma<256,256,3,1,2,2>(WPTR(10), U, W2);__syncthreads();
  ln_ip(W2, 256, 2, ra.ln_g[10], ra.ln_b[10], false);__syncthreads();
  conv_wmma<128,256,1,2,4,2>(WPTR(11), X, U);__syncthreads();   // downsample
  ln_ip(U, 256, 2, ra.ln_g[11], ra.ln_b[11], false); __syncthreads();
  combine_relu(X, W2, U, 256, 2);            __syncthreads();
  // ---- l3 block 1 (256,2) ----
  conv_wmma<256,256,3,1,2,2>(WPTR(12), X, U);__syncthreads();
  ln_ip(U, 256, 2, ra.ln_g[12], ra.ln_b[12], true);  __syncthreads();
  conv_wmma<256,256,3,1,2,2>(WPTR(13), U, W2);__syncthreads();
  ln_ip(W2, 256, 2, ra.ln_g[13], ra.ln_b[13], false);__syncthreads();
  combine_relu(X, W2, X, 256, 2);            __syncthreads();
#undef WPTR

  // mean over T=2 -> fp32 output (B,256)   X[s][t*256 + c]
  for (int p = tid; p < 16 * 256; p += RS_BT) {
    int s = p >> 8, c = p & 255;
    float v = 0.5f * ((float)X[s][c] + (float)X[s][256 + c]);
    ra.out[((size_t)blockIdx.x * 16 + s) * 256 + c] = v;
  }
}

// ===========================================================================
// host launcher
// ===========================================================================
extern "C" void kernel_launch(void* const* d_in, const int* in_sizes, int n_in,
                              void* d_out, int out_size, void* d_ws, size_t ws_size,
                              hipStream_t stream) {
  (void)in_sizes; (void)n_in; (void)out_size; (void)ws_size;
  // workspace: [ packed weights: 2008*512 halves | xfe: B*672 halves ]  ~24 MB
  _Float16* wpack = (_Float16*)d_ws;
  _Float16* xfe   = (_Float16*)((char*)d_ws + (size_t)kTotalFrags * 512 * sizeof(_Float16));

  auto fp = [&](int i) { return (const float*)d_in[i]; };

  // --- pack ResNet conv weights ---
  static const int wIdx[kNConv] = {28,31,34,37,40,43,46,49,52,55,58,61,64,67};
  PackArgs pa;
  for (int i = 0; i < kNConv; ++i) pa.w[i] = fp(wIdx[i]);
  pa.wpack = wpack;
  pack_weights_kernel<<<kTotalFrags, 32, 0, stream>>>(pa);

  // --- frontend ---
  FrontArgs fa;
  fa.pep_x = (const int*)d_in[0];
  fa.mhc_x = (const int*)d_in[1];
  fa.pep_emb = fp(2);  fa.mhc_emb = fp(3);
  fa.attn_in_w = fp(4);  fa.attn_in_b = fp(5);
  fa.attn_out_w = fp(6); fa.attn_out_b = fp(7);
  fa.cconv_w = fp(8); fa.cconv_b = fp(9);
  fa.cln_w = fp(10);  fa.cln_b = fp(11);
  for (int i = 0; i < 4; ++i) {
    fa.iconv_w[i] = fp(12 + 4 * i);
    fa.iconv_b[i] = fp(13 + 4 * i);
    fa.iln_w[i]   = fp(14 + 4 * i);
    fa.iln_b[i]   = fp(15 + 4 * i);
  }
  fa.xfe = xfe;
  frontend_kernel<<<BATCH, 32, 0, stream>>>(fa);

  // --- ResNet ---
  static const int lnIdx[kNConv][2] = {
    {29,30},{32,33},{35,36},{38,39},{41,42},{44,45},{47,48},
    {50,51},{53,54},{56,57},{59,60},{62,63},{65,66},{68,69}};
  ResArgs ra;
  ra.xfe = xfe;  ra.wpack = wpack;  ra.out = (float*)d_out;
  for (int i = 0; i < kNConv; ++i) { ra.ln_g[i] = fp(lnIdx[i][0]); ra.ln_b[i] = fp(lnIdx[i][1]); }
  resnet_kernel<<<BATCH / 16, RS_BT, 0, stream>>>(ra);
}
